// SignatureBlock_81037442941450
// MI455X (gfx1250) — matrix operand order repaired and checked by reading, not compiled, not run on hardware
//
#include <hip/hip_runtime.h>
#include <hip/hip_bf16.h>

typedef __attribute__((ext_vector_type(16))) _Float16 v16h;
typedef __attribute__((ext_vector_type(8)))  float    v8f;

#define SIG_K  992          // 31 + 31*31
#define BH_TOT 16384        // B*H = 1024*16
#define NOUT   1984
#define L_CH   31
#define E_STR  64
#define H_DIM  16

#define BM 128
#define BN 64
#define BK 32
#define NK (SIG_K / BK)     // 31 K-chunks

// ---------------- Kernel 1: W f32 -> f16, transposed (Wht[n][k] = W[k][n]) ---
__global__ void __launch_bounds__(256)
wconv_t_kernel(const float* __restrict__ W, _Float16* __restrict__ Wht) {
    __shared__ _Float16 tile[32][33];
    const int tx = threadIdx.x & 31;        // 0..31
    const int ty = threadIdx.x >> 5;        // 0..7
    const int n0 = blockIdx.x * 32;         // along NOUT (1984 = 62*32)
    const int k0 = blockIdx.y * 32;         // along SIG_K (992 = 31*32)

#pragma unroll
    for (int r = 0; r < 32; r += 8)
        tile[ty + r][tx] = (_Float16)W[(size_t)(k0 + ty + r) * NOUT + n0 + tx];
    __syncthreads();
#pragma unroll
    for (int r = 0; r < 32; r += 8)
        Wht[(size_t)(n0 + ty + r) * SIG_K + k0 + tx] = tile[tx][ty + r];
}

// ---------------- Kernel 2: depth-2 signature, one wave per batch element ----
__global__ void sig_kernel(const float* __restrict__ q, _Float16* __restrict__ sigh) {
    const int bh   = blockIdx.x;       // 0..16383
    const int b    = bh >> 4;
    const int h    = bh & 15;
    const int lane = threadIdx.x;      // 0..31

    __shared__ __align__(16) _Float16 AT[32][72]; // (p + 0.5*dx), [channel][stream]
    __shared__ __align__(16) _Float16 DT[32][72]; // dx

    {
        uint4 z = {0u, 0u, 0u, 0u};
#pragma unroll
        for (int c8 = 0; c8 < 8; ++c8) {
            *reinterpret_cast<uint4*>(&AT[lane][c8 * 8]) = z;
            *reinterpret_cast<uint4*>(&DT[lane][c8 * 8]) = z;
        }
    }

    if (lane < L_CH) {
        const float4* x4 = reinterpret_cast<const float4*>(
            q + (((size_t)b * L_CH + lane) * H_DIM + h) * E_STR);
        float prev = 0.f, p = 0.f;
#pragma unroll
        for (int c = 0; c < 16; ++c) {
            float4 v = x4[c];
            float vv[4] = {v.x, v.y, v.z, v.w};
#pragma unroll
            for (int i = 0; i < 4; ++i) {
                int e = c * 4 + i;
                if (e == 0) { prev = vv[0]; continue; }
                float dx = vv[i] - prev;
                AT[lane][e - 1] = (_Float16)(p + 0.5f * dx);
                DT[lane][e - 1] = (_Float16)dx;
                p += dx;
                prev = vv[i];
            }
        }
        sigh[(size_t)bh * SIG_K + lane] = (_Float16)p;   // s1
    }
    __syncthreads();

    const v8f zf = {0.f, 0.f, 0.f, 0.f, 0.f, 0.f, 0.f, 0.f};
    v8f acc[2][2];
    acc[0][0] = zf; acc[0][1] = zf; acc[1][0] = zf; acc[1][1] = zf;

    const int half16 = lane & 15;
    const int hi     = lane >> 4;

#pragma unroll
    for (int kc = 0; kc < 2; ++kc) {
        v16h aF[2], bF[2];
#pragma unroll
        for (int mi = 0; mi < 2; ++mi) {
            const int m  = mi * 16 + half16;
            const int kb = kc * 32 + hi * 8;
            uint4* ap = reinterpret_cast<uint4*>(&aF[mi]);
            ap[0] = *reinterpret_cast<const uint4*>(&AT[m][kb]);
            ap[1] = *reinterpret_cast<const uint4*>(&AT[m][kb + 16]);
        }
#pragma unroll
        for (int ni = 0; ni < 2; ++ni) {
            const int n  = ni * 16 + half16;
            const int k0 = kc * 32 + hi * 8;
            uint4* bp = reinterpret_cast<uint4*>(&bF[ni]);
            bp[0] = *reinterpret_cast<const uint4*>(&DT[n][k0]);
            bp[1] = *reinterpret_cast<const uint4*>(&DT[n][k0 + 16]);
        }
#pragma unroll
        for (int mi = 0; mi < 2; ++mi)
#pragma unroll
            for (int ni = 0; ni < 2; ++ni)
                acc[mi][ni] = __builtin_amdgcn_wmma_f32_16x16x32_f16(
                    false, aF[mi], false, bF[ni], (short)0, acc[mi][ni], false, false);
    }

    _Float16* srow = sigh + (size_t)bh * SIG_K;
#pragma unroll
    for (int mi = 0; mi < 2; ++mi)
#pragma unroll
        for (int ni = 0; ni < 2; ++ni)
#pragma unroll
            for (int r = 0; r < 8; ++r) {
                const int a = mi * 16 + (hi ? r + 8 : r);
                const int c = ni * 16 + half16;
                if (a < L_CH && c < L_CH)
                    srow[L_CH + a * L_CH + c] = (_Float16)acc[mi][ni][r];
            }
}

// ---------------- Kernel 3: sig @ W + b, double-buffered async-to-LDS --------
__device__ __forceinline__ void issue_tile_async(unsigned la0, unsigned la1, unsigned lb,
                                                 unsigned long long ga0,
                                                 unsigned long long gb0) {
    asm volatile("global_load_async_to_lds_b128 %0, %1, off"
                 :: "v"(la0), "v"(ga0) : "memory");
    asm volatile("global_load_async_to_lds_b128 %0, %1, off"
                 :: "v"(la1), "v"(ga0 + 16ull) : "memory");
    asm volatile("global_load_async_to_lds_b128 %0, %1, off"
                 :: "v"(lb), "v"(gb0) : "memory");
}

__global__ void __launch_bounds__(256)
gemm_kernel(const _Float16* __restrict__ sigh, const _Float16* __restrict__ Wht,
            const float* __restrict__ bias, float* __restrict__ out) {
    __shared__ __align__(16) _Float16 Asl[2][BM][40];  // double-buffered [m][k]
    __shared__ __align__(16) _Float16 Bsl[2][BN][40];  // double-buffered [n][k]

    const int tid    = threadIdx.x;
    const int tileN  = blockIdx.x;   // 0..30
    const int tileM  = blockIdx.y;   // 0..127
    const int lane   = tid & 31;
    const int w      = tid >> 5;     // 0..7
    const int wm     = w >> 1;       // 0..3
    const int wn     = w & 1;        // 0..1
    const int half16 = lane & 15;
    const int hi     = lane >> 4;

    const v8f zf = {0.f, 0.f, 0.f, 0.f, 0.f, 0.f, 0.f, 0.f};
    v8f acc[2][2];
    acc[0][0] = zf; acc[0][1] = zf; acc[1][0] = zf; acc[1][1] = zf;

    // staging maps: A = 128x32 halves -> 2 b128/thread; B = 64x32 halves -> 1 b128/thread
    const int rowA = tid >> 1;
    const int selA = tid & 1;
    const int nB   = tid >> 2;
    const int k0B  = (tid & 3) * 8;

    const _Float16* gA = sigh + (size_t)(tileM * BM + rowA) * SIG_K + selA * 16;
    const _Float16* gB = Wht + (size_t)(tileN * BN + nB) * SIG_K + k0B;

    // per-buffer LDS byte offsets (low 32 bits of flat LDS pointer == LDS offset)
    unsigned lA0[2], lA1[2], lB[2];
#pragma unroll
    for (int bfi = 0; bfi < 2; ++bfi) {
        lA0[bfi] = (unsigned)(size_t)&Asl[bfi][rowA][selA * 16];
        lA1[bfi] = lA0[bfi] + 16;
        lB[bfi]  = (unsigned)(size_t)&Bsl[bfi][nB][k0B];
    }

    // prologue: tile 0 -> buffer 0
    issue_tile_async(lA0[0], lA1[0], lB[0],
                     (unsigned long long)gA, (unsigned long long)gB);

    for (int kc = 0; kc < NK; ++kc) {
        const int cur = kc & 1;
        if (kc + 1 < NK) {
            // prefetch next tile into the other buffer (free since end of iter kc-1)
            const int nxt = cur ^ 1;
            issue_tile_async(lA0[nxt], lA1[nxt], lB[nxt],
                             (unsigned long long)(gA + (kc + 1) * BK),
                             (unsigned long long)(gB + (kc + 1) * BK));
            // async loads complete in order: <=3 outstanding => tile kc is resident
            asm volatile("s_wait_asynccnt 0x3" ::: "memory");
        } else {
            asm volatile("s_wait_asynccnt 0x0" ::: "memory");
        }
        __syncthreads();  // tile kc visible to all waves

        v16h aF[2], bF[2];
#pragma unroll
        for (int mi = 0; mi < 2; ++mi) {
            const int m  = wm * 32 + mi * 16 + half16;
            const int kb = hi * 8;
            uint4* ap = reinterpret_cast<uint4*>(&aF[mi]);
            ap[0] = *reinterpret_cast<const uint4*>(&Asl[cur][m][kb]);
            ap[1] = *reinterpret_cast<const uint4*>(&Asl[cur][m][kb + 16]);
        }
#pragma unroll
        for (int ni = 0; ni < 2; ++ni) {
            const int n  = wn * 32 + ni * 16 + half16;
            const int k0 = hi * 8;
            uint4* bp = reinterpret_cast<uint4*>(&bF[ni]);
            bp[0] = *reinterpret_cast<const uint4*>(&Bsl[cur][n][k0]);
            bp[1] = *reinterpret_cast<const uint4*>(&Bsl[cur][n][k0 + 16]);
        }
#pragma unroll
        for (int mi = 0; mi < 2; ++mi)
#pragma unroll
            for (int ni = 0; ni < 2; ++ni)
                acc[mi][ni] = __builtin_amdgcn_wmma_f32_16x16x32_f16(
                    false, aF[mi], false, bF[ni], (short)0, acc[mi][ni], false, false);

        __syncthreads();  // all waves done reading buf cur before it is refilled
    }

    // epilogue: bias + permuted scatter  out[b,o,h,e] = C[b*16+h, e*31+o]
    const int rbase = tileM * BM + wm * 32;
    const int cbase = tileN * BN + wn * 32;
#pragma unroll
    for (int ni = 0; ni < 2; ++ni) {
        const int col = cbase + ni * 16 + half16;      // < 1984
        const float bv = bias[col];
        const unsigned e = (unsigned)col / 31u;
        const unsigned o = (unsigned)col % 31u;
#pragma unroll
        for (int mi = 0; mi < 2; ++mi)
#pragma unroll
            for (int r = 0; r < 8; ++r) {
                const int row = rbase + mi * 16 + (hi ? r + 8 : r);  // b*16+h
                const int bb = row >> 4;
                const int hh = row & 15;
                out[(size_t)bb * 31744 + (size_t)o * 1024 + hh * 64 + e] =
                    acc[mi][ni][r] + bv;
            }
    }
}

extern "C" void kernel_launch(void* const* d_in, const int* in_sizes, int n_in,
                              void* d_out, int out_size, void* d_ws, size_t ws_size,
                              hipStream_t stream) {
    (void)in_sizes; (void)n_in; (void)out_size; (void)ws_size;
    const float* q  = (const float*)d_in[0];
    // d_in[1]=k, d_in[2]=v, d_in[3]=mask : unused by the reference
    const float* W  = (const float*)d_in[4];
    const float* bs = (const float*)d_in[5];
    float* out = (float*)d_out;

    _Float16* sigh = (_Float16*)d_ws;                       // 16384 x 992 f16
    _Float16* Wht  = sigh + (size_t)BH_TOT * SIG_K;         // 1984 x 992 f16 (W^T)

    wconv_t_kernel<<<dim3(NOUT / 32, SIG_K / 32), 256, 0, stream>>>(W, Wht);
    sig_kernel<<<BH_TOT, 32, 0, stream>>>(q, sigh);
    gemm_kernel<<<dim3(NOUT / BN, BH_TOT / BM), 256, 0, stream>>>(sigh, Wht, bs, out);
}
